// QuantumNATEnhanced_65481071397125
// MI455X (gfx1250) — matrix operand order repaired and checked by reading, not compiled, not run on hardware
//
#include <hip/hip_runtime.h>
#include <hip/hip_bf16.h>

#define BATCH 16384
#define NQ 10
#define NDEPTH 6
#define TWO_PI_F 6.283185307179586f

typedef __attribute__((ext_vector_type(16))) _Float16 v16h;
typedef __attribute__((ext_vector_type(8)))  float    v8f;

// LDS strides (padded to avoid bank conflicts)
#define PSI_STRIDE 1032   // halfs per psi row (1024 + 8)
#define PBUF_STRIDE 1028  // floats per p row (1024 + 4)

// ---------------------------------------------------------------------------
// Kernel 0: initialize reduction slots (must run every call; ws is not
// re-poisoned between graph replays).
// ---------------------------------------------------------------------------
__global__ void k_init(unsigned* __restrict__ gmaxSlot, float* __restrict__ sums) {
  if (threadIdx.x == 0) *gmaxSlot = 0u;          // ordered-uint 0 < every real float
  if (threadIdx.x < 8) sums[threadIdx.x] = 0.0f; // sum[4], sumsq[4]
}

// ---------------------------------------------------------------------------
// Kernel 1: 6x6 avg-pool of x[:, :, :24, :24] -> 16 values, encode to 10
// angles, global max via ordered-uint atomicMax. 4 images per 256-thr block.
// ---------------------------------------------------------------------------
__global__ void k_pool_encode(const float* __restrict__ x,
                              const float* __restrict__ encW,
                              const float* __restrict__ encB,
                              float* __restrict__ encOut,
                              unsigned* __restrict__ gmaxSlot)
{
  __shared__ float pooled[4][16];
  __shared__ float encL[4][10];
  unsigned tid = threadIdx.x;
  unsigned g = tid >> 6, l = tid & 63u;
  unsigned b = blockIdx.x * 4u + g;

  if (l < 16) {
    unsigned gy = l >> 2, gx = l & 3u;
    const float* base = x + (size_t)b * 784 + gy * 6 * 28 + gx * 6;
    float s = 0.0f;
#pragma unroll
    for (int iy = 0; iy < 6; ++iy)
#pragma unroll
      for (int ix = 0; ix < 6; ++ix) s += base[iy * 28 + ix];
    pooled[g][l] = s * (1.0f / 36.0f);
  }
  __syncthreads();
  if (l < 10) {
    float e = encB[l];
#pragma unroll
    for (int j = 0; j < 16; ++j) e += pooled[g][j] * encW[l * 16 + j];
    encOut[(size_t)b * 10 + l] = e;
    encL[g][l] = e;
  }
  __syncthreads();
  if (tid == 0) {
    float m = encL[0][0];
    for (int gg = 0; gg < 4; ++gg)
      for (int i = 0; i < 10; ++i) m = fmaxf(m, encL[gg][i]);
    unsigned u = __float_as_uint(m);
    unsigned ord = (u & 0x80000000u) ? ~u : (u | 0x80000000u); // monotone map
    atomicMax(gmaxSlot, ord);
  }
}

// ---------------------------------------------------------------------------
// Wave-resident 10-qubit state-vector primitives (wave32).
// Amp index k: bits[4:0] -> register r, bits[9:5] -> lane.
// ---------------------------------------------------------------------------
template <int Q>
__device__ __forceinline__ void gate1q(float (&sr)[32], float (&si)[32], unsigned lane,
                                       float g00r, float g00i, float g01r, float g01i,
                                       float g10r, float g10i, float g11r, float g11i)
{
  if constexpr (Q < 5) {
    constexpr int m = 1 << Q;
#pragma unroll
    for (int r = 0; r < 32; ++r) {
      if (!(r & m)) {
        const int r2 = r | m;
        float a0r = sr[r], a0i = si[r], a1r = sr[r2], a1i = si[r2];
        sr[r]  = g00r * a0r - g00i * a0i + g01r * a1r - g01i * a1i;
        si[r]  = g00r * a0i + g00i * a0r + g01r * a1i + g01i * a1r;
        sr[r2] = g10r * a0r - g10i * a0i + g11r * a1r - g11i * a1i;
        si[r2] = g10r * a0i + g10i * a0r + g11r * a1i + g11i * a1r;
      }
    }
  } else {
    constexpr int lm = 1 << (Q - 5);
    const bool hi = (lane & lm) != 0;
    const float ar = hi ? g10r : g00r, ai = hi ? g10i : g00i;
    const float br = hi ? g11r : g01r, bi = hi ? g11i : g01i;
#pragma unroll
    for (int r = 0; r < 32; ++r) {
      float pr = __shfl_xor(sr[r], lm, 32);
      float pi = __shfl_xor(si[r], lm, 32);
      float a0r = hi ? pr : sr[r], a0i = hi ? pi : si[r];
      float a1r = hi ? sr[r] : pr, a1i = hi ? si[r] : pi;
      sr[r] = ar * a0r - ai * a0i + br * a1r - bi * a1i;
      si[r] = ar * a0i + ai * a0r + br * a1i + bi * a1r;
    }
  }
}

template <int C, int T>
__device__ __forceinline__ void cnot(float (&sr)[32], float (&si)[32], unsigned lane)
{
  if constexpr (C < 5 && T < 5) {
    constexpr int cm = 1 << C, tm = 1 << T;
#pragma unroll
    for (int r = 0; r < 32; ++r) {
      if ((r & cm) && !(r & tm)) {
        const int r2 = r | tm;
        float t0 = sr[r]; sr[r] = sr[r2]; sr[r2] = t0;
        t0 = si[r]; si[r] = si[r2]; si[r2] = t0;
      }
    }
  } else if constexpr (C < 5 && T >= 5) {
    constexpr int cm = 1 << C, lm = 1 << (T - 5);
#pragma unroll
    for (int r = 0; r < 32; ++r) {
      if (r & cm) {
        sr[r] = __shfl_xor(sr[r], lm, 32);
        si[r] = __shfl_xor(si[r], lm, 32);
      }
    }
  } else if constexpr (C >= 5 && T < 5) {
    constexpr int lm = 1 << (C - 5), tm = 1 << T;
    const bool ctrl = (lane & lm) != 0;
#pragma unroll
    for (int r = 0; r < 32; ++r) {
      if (!(r & tm)) {
        const int r2 = r | tm;
        float a = sr[r], b = sr[r2];
        sr[r] = ctrl ? b : a; sr[r2] = ctrl ? a : b;
        a = si[r]; b = si[r2];
        si[r] = ctrl ? b : a; si[r2] = ctrl ? a : b;
      }
    }
  } else {
    constexpr int lm = 1 << (C - 5), tl = 1 << (T - 5);
    const bool ctrl = (lane & lm) != 0;
#pragma unroll
    for (int r = 0; r < 32; ++r) {
      float pr = __shfl_xor(sr[r], tl, 32);
      float pi = __shfl_xor(si[r], tl, 32);
      sr[r] = ctrl ? pr : sr[r];
      si[r] = ctrl ? pi : si[r];
    }
  }
}

// ---------------------------------------------------------------------------
// Kernel 2: build the fixed layer-unitary U (1024x1024 complex) as f16
// WT[n][k] = U[n,k] (per-output-column contiguous in k, so the GEMM's
// B-fragments are two contiguous b128 loads per lane). One wave per basis
// state k: full state in 64 VGPRs/lane, gates via register ops + shfl_xor.
// ---------------------------------------------------------------------------
__global__ void k_build_unitary(const float* __restrict__ weights,
                                _Float16* __restrict__ WTre,
                                _Float16* __restrict__ WTim)
{
  const unsigned lane = threadIdx.x & 31u;
  const unsigned k = blockIdx.x * 8u + (threadIdx.x >> 5); // 0..1023
  float sr[32], si[32];
#pragma unroll
  for (int r = 0; r < 32; ++r) {
    sr[r] = ((k >> 5) == lane && (k & 31u) == (unsigned)r) ? 1.0f : 0.0f;
    si[r] = 0.0f;
  }
  for (int d = 0; d < NDEPTH; ++d) {
    // G = RZ(az) @ RX(ax):
    // g00=(cz*ca,-sz*ca) g01=(-sz*sa,-cz*sa) g10=(sz*sa,-cz*sa) g11=(cz*ca,sz*ca)
#define LAYER_GATE(Q)                                                          \
    do {                                                                       \
      float ax = 0.5f * weights[(d * NQ + (Q)) * 2 + 0];                       \
      float az = 0.5f * weights[(d * NQ + (Q)) * 2 + 1];                       \
      float ca = cosf(ax), sa = sinf(ax), cz = cosf(az), sz = sinf(az);        \
      gate1q<(Q)>(sr, si, lane, cz * ca, -sz * ca, -sz * sa, -cz * sa,         \
                  sz * sa, -cz * sa, cz * ca, sz * ca);                        \
    } while (0)
    LAYER_GATE(0); LAYER_GATE(1); LAYER_GATE(2); LAYER_GATE(3); LAYER_GATE(4);
    LAYER_GATE(5); LAYER_GATE(6); LAYER_GATE(7); LAYER_GATE(8); LAYER_GATE(9);
#undef LAYER_GATE
    cnot<0, 1>(sr, si, lane); cnot<1, 2>(sr, si, lane); cnot<2, 3>(sr, si, lane);
    cnot<3, 4>(sr, si, lane); cnot<4, 5>(sr, si, lane); cnot<5, 6>(sr, si, lane);
    cnot<6, 7>(sr, si, lane); cnot<7, 8>(sr, si, lane); cnot<8, 9>(sr, si, lane);
    cnot<9, 0>(sr, si, lane);
  }
#pragma unroll
  for (int r = 0; r < 32; ++r) {
    size_t j = (size_t)lane * 32 + r;
    WTre[j * 1024 + k] = (_Float16)sr[r];
    WTim[j * 1024 + k] = (_Float16)si[r];
  }
}

// ---------------------------------------------------------------------------
// Kernel 3: fused GEMM kernel. Per 512-thread workgroup (16 waves): 32 batch
// rows x all 1024 output amps; each wave owns 64 columns (4 col-tiles) x the
// 2 row-blocks -> 16 v8f accumulators (128 VGPRs), no spills. PSI rows built
// in LDS from encoded angles, FINAL = PSI @ WT^T (complex) via
// v_wmma_f32_16x16x32_f16 (each B fragment feeds 4 WMMAs), then |.|^2,
// per-qubit <Z>, fc(10->4), batchnorm partial sums via f32 atomics.
// ---------------------------------------------------------------------------
__global__ __launch_bounds__(512, 1)
void k_qgemm(const float* __restrict__ encoded,
             const unsigned* __restrict__ gmaxSlot,
             const float* __restrict__ fcW,
             const float* __restrict__ fcb,
             const _Float16* __restrict__ WTre,
             const _Float16* __restrict__ WTim,
             float* __restrict__ outRaw,
             float* __restrict__ sums)
{
  extern __shared__ char smemRaw[];
  float*    pbuf  = (float*)smemRaw;              // 32*PBUF_STRIDE floats (p values)
  _Float16* psi   = (_Float16*)smemRaw;           // 32*PSI_STRIDE halfs (overlaps pbuf; dead before p-write)
  float*    zpart = pbuf + 32 * PBUF_STRIDE;      // 32*16*7
  float*    zrow  = zpart + 32 * 16 * 7;          // 320
  float*    outL  = zrow + 320;                   // 128
  float*    csL   = outL + 128;                   // 32*10*2

  const unsigned tid  = threadIdx.x;
  const unsigned lane = tid & 31u;
  const unsigned wave = tid >> 5;       // 0..15, each owns 64 columns
  const unsigned h    = lane >> 4;      // K-half selector within fragments
  const unsigned nlo  = lane & 15u;     // row (A) or column (B/C) id
  const unsigned rowbase = blockIdx.x * 32u;

  // decode ordered-uint global max
  unsigned gmo = *gmaxSlot;
  float gmax = (gmo & 0x80000000u) ? __uint_as_float(gmo ^ 0x80000000u)
                                   : __uint_as_float(~gmo);
  const float scale = TWO_PI_F / (gmax + 1e-8f);

  // cos/sin of half-angles for the 32 rows x 10 wires
  for (unsigned o = tid; o < 320; o += 512) {
    unsigned row = o / 10u, i = o % 10u;
    float e = encoded[(size_t)(rowbase + row) * 10 + i] * scale * 0.5f;
    csL[o * 2 + 0] = cosf(e);
    csL[o * 2 + 1] = sinf(e);
  }
  __syncthreads();

  // build PSI rows in LDS: psi[row][k] = prod_i (bit_i(k) ? s_i : c_i)
  {
    unsigned row = tid >> 4, seg = tid & 15u;   // 32 rows x 16 segments of 64
    float c[10], s[10];
#pragma unroll
    for (int i = 0; i < 10; ++i) {
      c[i] = csL[(row * 10 + i) * 2 + 0];
      s[i] = csL[(row * 10 + i) * 2 + 1];
    }
    for (int u = 0; u < 64; ++u) {
      unsigned kk = seg * 64u + u;
      float v = 1.0f;
#pragma unroll
      for (int i = 0; i < 10; ++i) v *= ((kk >> i) & 1u) ? s[i] : c[i];
      psi[(size_t)row * PSI_STRIDE + kk] = (_Float16)v;
    }
  }
  __syncthreads();

  // --- main WMMA GEMM: FINAL[32 x 1024] (complex) ---
  v8f accRe0[4], accIm0[4], accRe1[4], accIm1[4];
  {
    v8f vz = {};
#pragma unroll
    for (int t = 0; t < 4; ++t) { accRe0[t] = vz; accIm0[t] = vz; accRe1[t] = vz; accIm1[t] = vz; }
  }

  union Frag { v16h v; float4 f4[2]; };
  const _Float16* psiB0 = psi + (size_t)nlo * PSI_STRIDE;
  const _Float16* psiB1 = psi + (size_t)(nlo + 16) * PSI_STRIDE;

  for (int ks = 0; ks < 32; ++ks) {
    Frag A0, A1;
    // A layout (16-bit A 16x32): halfs 0-7 -> K=8h+e ; halfs 8-15 -> K=16+8h+e
    const _Float16* pA0 = psiB0 + ks * 32 + 8 * h;
    A0.f4[0] = *(const float4*)(pA0);
    A0.f4[1] = *(const float4*)(pA0 + 16);
    const _Float16* pA1 = psiB1 + ks * 32 + 8 * h;
    A1.f4[0] = *(const float4*)(pA1);
    A1.f4[1] = *(const float4*)(pA1 + 16);
#pragma unroll
    for (int t = 0; t < 4; ++t) {
      size_t n = (size_t)wave * 64 + t * 16 + nlo;
      // B layout (16-bit B 32x16): halfs 0-7 -> K=8h+e ; halfs 8-15 -> K=16+8h+e
      const _Float16* pr = WTre + n * 1024 + ks * 32 + 8 * h;
      const _Float16* pi = WTim + n * 1024 + ks * 32 + 8 * h;
      Frag Bre, Bim;
      Bre.f4[0] = *(const float4*)(pr);
      Bre.f4[1] = *(const float4*)(pr + 16);
      Bim.f4[0] = *(const float4*)(pi);
      Bim.f4[1] = *(const float4*)(pi + 16);
      accRe0[t] = __builtin_amdgcn_wmma_f32_16x16x32_f16(false, A0.v, false, Bre.v, (short)0, accRe0[t], false, false);
      accIm0[t] = __builtin_amdgcn_wmma_f32_16x16x32_f16(false, A0.v, false, Bim.v, (short)0, accIm0[t], false, false);
      accRe1[t] = __builtin_amdgcn_wmma_f32_16x16x32_f16(false, A1.v, false, Bre.v, (short)0, accRe1[t], false, false);
      accIm1[t] = __builtin_amdgcn_wmma_f32_16x16x32_f16(false, A1.v, false, Bim.v, (short)0, accIm1[t], false, false);
    }
  }
  __syncthreads(); // psi dead; reuse region as pbuf

  // p = re^2 + im^2  (C layout: lane&15 = col, vgpr r -> row r + 8h)
#pragma unroll
  for (int t = 0; t < 4; ++t) {
    unsigned col = wave * 64u + t * 16u + nlo;
#pragma unroll
    for (int r = 0; r < 8; ++r) {
      unsigned row0 = (unsigned)r + 8u * h;
      float re = accRe0[t][r], im = accIm0[t][r];
      pbuf[row0 * PBUF_STRIDE + col] = re * re + im * im;
      re = accRe1[t][r]; im = accIm1[t][r];
      pbuf[(row0 + 16u) * PBUF_STRIDE + col] = re * re + im * im;
    }
  }
  __syncthreads();

  // chunk-wise signed reductions: 64 cols per chunk; bits 0..5 vary in-chunk,
  // bits 6..9 are the chunk id. 512 threads = 32 rows x 16 chunks.
  {
    unsigned row = tid >> 4, chunk = tid & 15u;
    float tot = 0.0f, zq[6] = {0, 0, 0, 0, 0, 0};
    for (int u = 0; u < 64; ++u) {
      unsigned jj = chunk * 64u + (((unsigned)u + chunk) & 63u); // bank stagger
      float p = pbuf[row * PBUF_STRIDE + jj];
      tot += p;
#pragma unroll
      for (int q = 0; q < 6; ++q)
        zq[q] += __uint_as_float(__float_as_uint(p) ^ (((jj >> q) & 1u) << 31));
    }
    float* zp = zpart + (row * 16u + chunk) * 7u;
    zp[0] = tot;
#pragma unroll
    for (int q = 0; q < 6; ++q) zp[1 + q] = zq[q];
  }
  __syncthreads();

  // combine chunks -> z[row][qubit]
  for (unsigned o = tid; o < 320; o += 512) {
    unsigned row = o / 10u, q = o % 10u;
    float acc = 0.0f;
#pragma unroll
    for (int ch = 0; ch < 16; ++ch) {
      const float* zp = zpart + (row * 16u + (unsigned)ch) * 7u;
      if (q < 6) acc += zp[1 + q];
      else acc += (((unsigned)ch >> (q - 6)) & 1u) ? -zp[0] : zp[0];
    }
    zrow[o] = acc;
  }
  __syncthreads();

  // fc: out[b,c] = z . fcW[c,:] + fcb[c]
  if (tid < 128) {
    unsigned row = tid >> 2, c = tid & 3u;
    float o = fcb[c];
#pragma unroll
    for (int i = 0; i < 10; ++i) o += zrow[row * 10u + i] * fcW[c * 10 + i];
    outRaw[(size_t)(rowbase + row) * 4 + c] = o;
    outL[row * 4u + c] = o;
  }
  __syncthreads();
  if (tid < 4) {
    float s = 0.0f, s2 = 0.0f;
#pragma unroll
    for (int r = 0; r < 32; ++r) { float o = outL[(unsigned)r * 4u + tid]; s += o; s2 += o * o; }
    atomicAdd(&sums[tid], s);
    atomicAdd(&sums[4 + tid], s2);
  }
}

// ---------------------------------------------------------------------------
// Kernel 4: batchnorm finalize
// ---------------------------------------------------------------------------
__global__ void k_bn(const float* __restrict__ outRaw,
                     const float* __restrict__ sums,
                     const float* __restrict__ gamma,
                     const float* __restrict__ beta,
                     float* __restrict__ out)
{
  unsigned idx = blockIdx.x * 256u + threadIdx.x;
  if (idx >= (unsigned)BATCH * 4u) return;
  unsigned c = idx & 3u;
  float mean = sums[c] * (1.0f / (float)BATCH);
  float var = sums[4 + c] * (1.0f / (float)BATCH) - mean * mean;
  float inv = rsqrtf(var + 1e-5f);
  out[idx] = gamma[c] * (outRaw[idx] - mean) * inv + beta[c];
}

// ---------------------------------------------------------------------------
extern "C" void kernel_launch(void* const* d_in, const int* in_sizes, int n_in,
                              void* d_out, int out_size, void* d_ws, size_t ws_size,
                              hipStream_t stream) {
  const float* x    = (const float*)d_in[0];
  const float* encW = (const float*)d_in[1];
  const float* encB = (const float*)d_in[2];
  const float* wts  = (const float*)d_in[3];
  const float* fcW  = (const float*)d_in[4];
  const float* fcb  = (const float*)d_in[5];
  const float* bg   = (const float*)d_in[6];
  const float* bb   = (const float*)d_in[7];
  float* out = (float*)d_out;

  // workspace layout
  char* ws = (char*)d_ws;
  unsigned* gmaxSlot = (unsigned*)ws;                 // 4B
  float* sums   = (float*)(ws + 16);                  // 8 floats
  float* enc    = (float*)(ws + 64);                  // BATCH*10
  float* outRaw = enc + (size_t)BATCH * 10;           // BATCH*4
  size_t off = 64 + ((size_t)BATCH * 10 + (size_t)BATCH * 4) * sizeof(float);
  off = (off + 255) & ~(size_t)255;
  _Float16* WTre = (_Float16*)(ws + off);             // 1024*1024 halfs (2 MB)
  _Float16* WTim = WTre + (size_t)1024 * 1024;        // 2 MB

  k_init<<<1, 32, 0, stream>>>(gmaxSlot, sums);
  k_pool_encode<<<BATCH / 4, 256, 0, stream>>>(x, encW, encB, enc, gmaxSlot);
  k_build_unitary<<<128, 256, 0, stream>>>(wts, WTre, WTim);

  size_t smem = (size_t)(32 * PBUF_STRIDE + 32 * 16 * 7 + 320 + 128 + 32 * 10 * 2) * sizeof(float);
  k_qgemm<<<BATCH / 32, 512, smem, stream>>>(enc, gmaxSlot, fcW, fcb, WTre, WTim, outRaw, sums);

  k_bn<<<(BATCH * 4 + 255) / 256, 256, 0, stream>>>(outRaw, sums, bg, bb, out);
}
